// DecLayer_13262859010364
// MI455X (gfx1250) — matrix operand order repaired and checked by reading, not compiled, not run on hardware
//
#include <hip/hip_runtime.h>
#include <hip/hip_bf16.h>
#include <math.h>

// ---------------------------------------------------------------------------
// Problem constants (match reference)
// ---------------------------------------------------------------------------
#define H    128          // hidden
#define NIN  384          // edge feature dim
#define CIN  512          // H + NIN
#define FF   512          // 4*H
#define KNB  48           // neighbors per node
#define EPSF 1e-5f
#define SCALE_INV (1.0f / 30.0f)

// ---------------------------------------------------------------------------
// WMMA types (CDNA5 / gfx1250, wave32)
// ---------------------------------------------------------------------------
typedef __attribute__((ext_vector_type(16))) __bf16 v16bf;
typedef __attribute__((ext_vector_type(8)))  __bf16 v8bf;
typedef __attribute__((ext_vector_type(8)))  float  v8f;

__device__ __forceinline__ v8f wmma_bf16(v16bf a, v16bf b, v8f c) {
    // D = A(16x32 bf16) * B(32x16 bf16) + C(16x16 f32)
    return __builtin_amdgcn_wmma_f32_16x16x32_bf16(
        /*neg_a=*/false, a, /*neg_b=*/false, b,
        /*c_mod=*/(short)0, c, /*reuse_a=*/false, /*reuse_b=*/false);
}

__device__ __forceinline__ v16bf combine8(v8bf lo, v8bf hi) {
    return __builtin_shufflevector(lo, hi, 0,1,2,3,4,5,6,7,8,9,10,11,12,13,14,15);
}

// A fragment (16x32): lane = M row; lanes 0-15 hold K [kb+0..7] and [kb+16..23],
// lanes 16-31 hold K [kb+8..15] and [kb+24..31].  base is row-major [row][k].
__device__ __forceinline__ v16bf load_a_frag(const __bf16* base, int stride,
                                             int row, int kb, int klh) {
    const __bf16* p = base + (size_t)row * stride + kb + klh * 8;
    v8bf lo = *(const v8bf*)(p);
    v8bf hi = *(const v8bf*)(p + 16);
    return combine8(lo, hi);
}

// B fragment (32x16): lane = N col; lanes 0-15 hold K [kb+0..7] and [kb+16..23],
// lanes 16-31 hold K [kb+8..15] and [kb+24..31].  wt is pre-transposed [n][k] bf16.
__device__ __forceinline__ v16bf load_b_frag(const __bf16* wt, int Kd,
                                             int n, int kb, int klh) {
    const __bf16* p = wt + (size_t)n * Kd + kb + klh * 8;
    v8bf lo = *(const v8bf*)(p);
    v8bf hi = *(const v8bf*)(p + 16);
    return combine8(lo, hi);
}

__device__ __forceinline__ float gelu_f(float x) {
    return 0.5f * x * (1.0f + erff(x * 0.70710678118654752f));
}

// ---------------------------------------------------------------------------
// Weight pre-pass: fp32 [k][n] row-major  ->  bf16 [n][k] (transposed)
// ---------------------------------------------------------------------------
__global__ void k_transpose(const float* __restrict__ W, __bf16* __restrict__ Wt,
                            int Kd, int Nd) {
    int i = blockIdx.x * blockDim.x + threadIdx.x;
    if (i < Kd * Nd) {
        int n = i / Kd;
        int k = i % Kd;
        Wt[i] = (__bf16)W[(size_t)k * Nd + n];
    }
}

// ---------------------------------------------------------------------------
// Kernel 1: edge MLP + masked neighbor reduction.  One block per node (B*N).
//   rows = 48 neighbors, GEMM1: 48x512 @ 512x128, GEMM2/3: 48x128 @ 128x128
// ---------------------------------------------------------------------------
__global__ __launch_bounds__(256)
void k_edge_mlp(const float* __restrict__ hV, const float* __restrict__ hE,
                const float* __restrict__ maskAtt,
                const __bf16* __restrict__ W1t, const float* __restrict__ B1,
                const __bf16* __restrict__ W2t, const float* __restrict__ B2,
                const __bf16* __restrict__ W3t, const float* __restrict__ B3,
                float* __restrict__ dh) {
    __shared__ __align__(16) __bf16 sA[KNB * CIN];   // 48 KB: h_EV, later A2/A3
    __shared__ float sMask[KNB];

    const int node = blockIdx.x;
    const int t    = threadIdx.x;
    const int wave = t >> 5;
    const int lane = t & 31;
    const int nlane = lane & 15;     // N column within tile
    const int klh   = lane >> 4;     // lane-half (K split)
    const int ncol  = wave * 16 + nlane;   // this lane's output column (0..127)

    // ---- stage A = [h_V | h_E] as bf16 into LDS (48x512) -------------------
    const float* hVrow = hV + (size_t)node * H;
    const float* hErow = hE + (size_t)node * KNB * NIN;
    for (int q = t; q < (KNB * CIN) / 4; q += 256) {
        int e = q * 4;
        int r = e / CIN, c = e % CIN;
        float4 v = (c < H) ? *(const float4*)(hVrow + c)
                           : *(const float4*)(hErow + (size_t)r * NIN + (c - H));
        sA[e + 0] = (__bf16)v.x; sA[e + 1] = (__bf16)v.y;
        sA[e + 2] = (__bf16)v.z; sA[e + 3] = (__bf16)v.w;
    }
    if (t < KNB) sMask[t] = maskAtt[(size_t)node * KNB + t];
    __builtin_prefetch(W1t + (size_t)ncol * CIN, 0, 3);   // global_prefetch_b8
    __syncthreads();

    // ---- GEMM1: (48x512)x(512x128), 3 M-tiles per wave ---------------------
    v8f acc[3];
    #pragma unroll
    for (int mt = 0; mt < 3; ++mt) acc[mt] = (v8f){0,0,0,0,0,0,0,0};
    for (int kt = 0; kt < CIN / 32; ++kt) {
        int kb = kt * 32;
        v16bf bv = load_b_frag(W1t, CIN, ncol, kb, klh);
        #pragma unroll
        for (int mt = 0; mt < 3; ++mt) {
            v16bf av = load_a_frag(sA, CIN, mt * 16 + nlane, kb, klh);
            acc[mt] = wmma_bf16(av, bv, acc[mt]);
        }
    }
    __syncthreads();                     // all reads of h_EV done
    {
        float bias = B1[ncol];           // A2 = gelu(C1+b1), 48x128 bf16 at sA[0]
        #pragma unroll
        for (int mt = 0; mt < 3; ++mt)
            #pragma unroll
            for (int r = 0; r < 8; ++r) {
                int m = mt * 16 + klh * 8 + r;
                sA[(size_t)m * H + ncol] = (__bf16)gelu_f(acc[mt][r] + bias);
            }
    }
    __syncthreads();

    // ---- GEMM2: (48x128)x(128x128) ----------------------------------------
    #pragma unroll
    for (int mt = 0; mt < 3; ++mt) acc[mt] = (v8f){0,0,0,0,0,0,0,0};
    #pragma unroll
    for (int kt = 0; kt < H / 32; ++kt) {
        int kb = kt * 32;
        v16bf bv = load_b_frag(W2t, H, ncol, kb, klh);
        #pragma unroll
        for (int mt = 0; mt < 3; ++mt) {
            v16bf av = load_a_frag(sA, H, mt * 16 + nlane, kb, klh);
            acc[mt] = wmma_bf16(av, bv, acc[mt]);
        }
    }
    __bf16* sA3 = sA + KNB * H;          // disjoint from A2 region
    {
        float bias = B2[ncol];
        #pragma unroll
        for (int mt = 0; mt < 3; ++mt)
            #pragma unroll
            for (int r = 0; r < 8; ++r) {
                int m = mt * 16 + klh * 8 + r;
                sA3[(size_t)m * H + ncol] = (__bf16)gelu_f(acc[mt][r] + bias);
            }
    }
    __syncthreads();

    // ---- GEMM3 + mask + reduce over 48 neighbors --------------------------
    #pragma unroll
    for (int mt = 0; mt < 3; ++mt) acc[mt] = (v8f){0,0,0,0,0,0,0,0};
    #pragma unroll
    for (int kt = 0; kt < H / 32; ++kt) {
        int kb = kt * 32;
        v16bf bv = load_b_frag(W3t, H, ncol, kb, klh);
        #pragma unroll
        for (int mt = 0; mt < 3; ++mt) {
            v16bf av = load_a_frag(sA3, H, mt * 16 + nlane, kb, klh);
            acc[mt] = wmma_bf16(av, bv, acc[mt]);
        }
    }
    {
        float bias = B3[ncol];
        float csum = 0.f;
        #pragma unroll
        for (int mt = 0; mt < 3; ++mt)
            #pragma unroll
            for (int r = 0; r < 8; ++r) {
                int m = mt * 16 + klh * 8 + r;
                csum += (acc[mt][r] + bias) * sMask[m];
            }
        csum += __shfl_down(csum, 16);   // fold upper lane-half (same column)
        if (lane < 16)
            dh[(size_t)node * H + wave * 16 + lane] = csum * SCALE_INV;
    }
}

// ---------------------------------------------------------------------------
// half-wave LayerNorm statistics: wave w -> rows 2w, 2w+1 (16 lanes each)
// ---------------------------------------------------------------------------
__device__ __forceinline__ void row_stats(const float* buf, float* sMu, float* sRs,
                                          int wave, int lane) {
    int row = wave * 2 + (lane >> 4);
    int j   = lane & 15;
    float s = 0.f, s2 = 0.f;
    #pragma unroll
    for (int c = j; c < H; c += 16) { float v = buf[row * H + c]; s += v; s2 += v * v; }
    #pragma unroll
    for (int m = 8; m >= 1; m >>= 1) { s += __shfl_xor(s, m); s2 += __shfl_xor(s2, m); }
    if (j == 0) {
        float mu = s * (1.0f / H);
        sMu[row] = mu;
        sRs[row] = rsqrtf(s2 * (1.0f / H) - mu * mu + EPSF);
    }
}

// ---------------------------------------------------------------------------
// Kernel 2: LN1 -> FFN (128->512 gelu ->128) -> residual -> LN2 -> mask_V
//   16 nodes per block.
// ---------------------------------------------------------------------------
__global__ __launch_bounds__(256)
void k_node_ffn(const float* __restrict__ hV, const float* __restrict__ dh,
                const float* __restrict__ maskV,
                const __bf16* __restrict__ Wint,  const float* __restrict__ Bin,
                const __bf16* __restrict__ Woutt, const float* __restrict__ Bout,
                const float* __restrict__ g1, const float* __restrict__ bb1,
                const float* __restrict__ g2, const float* __restrict__ bb2,
                float* __restrict__ out) {
    __shared__ __align__(16) float  sX[16 * H];    // x, later y   (8 KB)
    __shared__ __align__(16) float  sH[16 * H];    // LN1 output   (8 KB)
    __shared__ __align__(16) __bf16 sAbf[16 * H];  // LN1 bf16     (4 KB)
    __shared__ __align__(16) __bf16 sG[16 * FF];   // gelu(FFN1)  (16 KB)
    __shared__ float sMu[16], sRs[16];

    const int tile = blockIdx.x;
    const int t    = threadIdx.x;
    const int wave = t >> 5;
    const int lane = t & 31;
    const int nlane = lane & 15;
    const int klh   = lane >> 4;
    const size_t base = (size_t)tile * 16 * H;

    // x = h_V + dh
    for (int i = t; i < 16 * H; i += 256) sX[i] = hV[base + i] + dh[base + i];
    __syncthreads();
    row_stats(sX, sMu, sRs, wave, lane);
    __syncthreads();
    // h = LN1(x);  keep fp32 residual + bf16 GEMM operand
    for (int i = t; i < 16 * H; i += 256) {
        int r = i / H, c = i % H;
        float hn = (sX[i] - sMu[r]) * sRs[r] * g1[c] + bb1[c];
        sH[i]   = hn;
        sAbf[i] = (__bf16)hn;
    }
    __syncthreads();

    // ---- FFN1: (16x128)x(128x512); wave owns 4 N-tiles --------------------
    v8f acc[4];
    #pragma unroll
    for (int q = 0; q < 4; ++q) acc[q] = (v8f){0,0,0,0,0,0,0,0};
    #pragma unroll
    for (int kt = 0; kt < H / 32; ++kt) {
        int kb = kt * 32;
        v16bf av = load_a_frag(sAbf, H, nlane, kb, klh);
        #pragma unroll
        for (int q = 0; q < 4; ++q) {
            int n = (wave * 4 + q) * 16 + nlane;
            v16bf bv = load_b_frag(Wint, H, n, kb, klh);
            acc[q] = wmma_bf16(av, bv, acc[q]);
        }
    }
    #pragma unroll
    for (int q = 0; q < 4; ++q) {
        int n = (wave * 4 + q) * 16 + nlane;
        float bias = Bin[n];
        #pragma unroll
        for (int r = 0; r < 8; ++r) {
            int m = klh * 8 + r;
            sG[(size_t)m * FF + n] = (__bf16)gelu_f(acc[q][r] + bias);
        }
    }
    __syncthreads();

    // ---- FFN2: (16x512)x(512x128); wave owns 1 N-tile ---------------------
    v8f c2 = (v8f){0,0,0,0,0,0,0,0};
    const int n2 = wave * 16 + nlane;
    for (int kt = 0; kt < FF / 32; ++kt) {
        int kb = kt * 32;
        v16bf av = load_a_frag(sG, FF, nlane, kb, klh);
        v16bf bv = load_b_frag(Woutt, FF, n2, kb, klh);
        c2 = wmma_bf16(av, bv, c2);
    }
    {
        float bias = Bout[n2];
        #pragma unroll
        for (int r = 0; r < 8; ++r) {
            int m = klh * 8 + r;
            sX[(size_t)m * H + n2] = c2[r] + bias + sH[(size_t)m * H + n2];  // y
        }
    }
    __syncthreads();
    row_stats(sX, sMu, sRs, wave, lane);
    __syncthreads();
    // out = mask_V * LN2(y)
    for (int i = t; i < 16 * H; i += 256) {
        int r = i / H, c = i % H;
        float y = (sX[i] - sMu[r]) * sRs[r] * g2[c] + bb2[c];
        out[base + i] = maskV[(size_t)tile * 16 + r] * y;
    }
}

// ---------------------------------------------------------------------------
// Host launcher
// ---------------------------------------------------------------------------
extern "C" void kernel_launch(void* const* d_in, const int* in_sizes, int n_in,
                              void* d_out, int out_size, void* d_ws, size_t ws_size,
                              hipStream_t stream) {
    const float* hV      = (const float*)d_in[0];
    const float* hE      = (const float*)d_in[1];
    const float* maskV   = (const float*)d_in[2];
    const float* maskAtt = (const float*)d_in[3];
    const float* W1 = (const float*)d_in[4];   const float* B1 = (const float*)d_in[5];
    const float* W2 = (const float*)d_in[6];   const float* B2 = (const float*)d_in[7];
    const float* W3 = (const float*)d_in[8];   const float* B3 = (const float*)d_in[9];
    const float* Win  = (const float*)d_in[10]; const float* Bin  = (const float*)d_in[11];
    const float* Wout = (const float*)d_in[12]; const float* Bout = (const float*)d_in[13];
    const float* g1 = (const float*)d_in[14];  const float* bb1 = (const float*)d_in[15];
    const float* g2 = (const float*)d_in[16];  const float* bb2 = (const float*)d_in[17];

    const int BN = in_sizes[0] / H;            // B*N nodes (8192)

    // workspace layout: transposed bf16 weights (~448 KB) then dh (4 MB)
    __bf16* W1t   = (__bf16*)d_ws;             // [H][CIN]
    __bf16* W2t   = W1t + (size_t)H * CIN;     // [H][H]
    __bf16* W3t   = W2t + (size_t)H * H;       // [H][H]
    __bf16* Wint  = W3t + (size_t)H * H;       // [FF][H]
    __bf16* Woutt = Wint + (size_t)FF * H;     // [H][FF]
    float*  dh    = (float*)(Woutt + (size_t)H * FF);

    auto T = [&](const float* W, __bf16* Wt, int Kd, int Nd) {
        int total = Kd * Nd;
        k_transpose<<<(total + 255) / 256, 256, 0, stream>>>(W, Wt, Kd, Nd);
    };
    T(W1,   W1t,   CIN, H);
    T(W2,   W2t,   H,   H);
    T(W3,   W3t,   H,   H);
    T(Win,  Wint,  H,   FF);
    T(Wout, Woutt, FF,  H);

    k_edge_mlp<<<BN, 256, 0, stream>>>(hV, hE, maskAtt,
                                       W1t, B1, W2t, B2, W3t, B3, dh);
    k_node_ffn<<<BN / 16, 256, 0, stream>>>(hV, dh, maskV,
                                            Wint, Bin, Woutt, Bout,
                                            g1, bb1, g2, bb2, (float*)d_out);
}
